// EfficientCrossAttention2D_16750372454958
// MI455X (gfx1250) — matrix operand order repaired and checked by reading, not compiled, not run on hardware
//
#include <hip/hip_runtime.h>

// ---------------------------------------------------------------------------
// EfficientCrossAttention2D for MI455X (gfx1250, wave32, WMMA)
// n=16, C=256, H=W=96, L=9216, HEADS=8, head_dim=32
// ---------------------------------------------------------------------------

#define N_BATCH 16
#define C_DIM   256
#define L_DIM   9216
#define NHEADS  8

typedef __attribute__((ext_vector_type(16))) __bf16 v16bf;
typedef __attribute__((ext_vector_type(8)))  __bf16 v8bf;
typedef __attribute__((ext_vector_type(2)))  __bf16 v2bf;
typedef __attribute__((ext_vector_type(8)))  float  v8f;
typedef __attribute__((ext_vector_type(2)))  float  v2f;

__device__ __forceinline__ v8f wmma_bf16(v16bf a, v16bf b, v8f c) {
  // (neg_a, A, neg_b, B, c_mod, C, reuse_a, reuse_b)
  return __builtin_amdgcn_wmma_f32_16x16x32_bf16(false, a, false, b, (short)0, c,
                                                 false, false);
}

// ---------------------------------------------------------------------------
// Kernel 1: fused K+V projection.
//   K[b,o,l] = sum_c Wk[o,c]*X[b,c,l] + bk[o]
//   V[b,o,l] = sum_c Wv[o,c]*X[b,c,l] + bv[o]
// Both GEMMs share the B operand (context), so each workgroup stages its
// 32x128 context tile into LDS ONCE and runs it against both weight tiles:
// halves context HBM traffic, 8 WMMA per wave per K-step.
// Tile: 64(M) x 128(N), K-steps of 32, 8 waves. grid = (L/128, C/64, n)
// ---------------------------------------------------------------------------
__global__ __launch_bounds__(256) void kv_proj_kernel(
    const float* __restrict__ Wk, const float* __restrict__ bk,
    const float* __restrict__ Wv, const float* __restrict__ bv,
    const float* __restrict__ X,
    __bf16* __restrict__ OutK, __bf16* __restrict__ OutV)
{
  __shared__ __attribute__((aligned(32))) __bf16 ldsA[8 * 512];  // [0..3]=Wk, [4..7]=Wv
  __shared__ __attribute__((aligned(32))) __bf16 ldsB[8 * 512];  // 8 subtiles 32x16

  const int tid  = threadIdx.x;
  const int wave = tid >> 5, lane = tid & 31;
  const int half = lane >> 4, ln = lane & 15;
  const int mi = wave & 3;             // A subtile row for this wave
  const int nb = (wave >> 2) << 2;     // first of four B subtiles
  const int m0 = blockIdx.y * 64;
  const int l0 = blockIdx.x * 128;
  const size_t XB = (size_t)blockIdx.z * C_DIM * L_DIM;

  v8f accK[4] = {{}, {}, {}, {}};
  v8f accV[4] = {{}, {}, {}, {}};

  for (int k0 = 0; k0 < C_DIM; k0 += 32) {
    // --- stage both A tiles (Wk, Wv 64x32): b64 loads, packed b32 stores ---
#pragma unroll
    for (int i = 0; i < 8; i++) {
      int idx = tid + i * 256;                        // 2048 pairs total
      int which = idx >> 10;                          // 0 = Wk, 1 = Wv
      int pidx  = idx & 1023;
      int k2 = pidx & 15, m = pidx >> 4;              // k = 2*k2 (even)
      const float* src = which ? Wv : Wk;
      v2f w2 = *(const v2f*)&src[(size_t)(m0 + m) * C_DIM + (k0 + 2 * k2)];
      int k  = 2 * k2;
      int hh = k >> 4;                                // A frag: e -> k=half*16+e
      int e  = k & 15;
      v2bf p; p[0] = (__bf16)w2[0]; p[1] = (__bf16)w2[1];
      *(v2bf*)&ldsA[(((which << 2) + (m >> 4)) << 9) + (((hh << 4) + (m & 15)) << 4) + e] = p;
    }
    // --- stage shared B tile (32x128 of X): pack over adjacent k rows ---
#pragma unroll
    for (int i = 0; i < 8; i++) {
      int idx = tid + i * 256;                        // 2048 pairs
      int nn = idx & 127, k2 = idx >> 7;              // k = 2*k2
      int k  = 2 * k2;
      const float* xp = &X[XB + (size_t)(k0 + k) * L_DIM + (l0 + nn)];
      float a0 = xp[0], a1 = xp[L_DIM];               // rows k, k+1 (coalesced)
      if (k0 + 32 < C_DIM) __builtin_prefetch(xp + 32 * L_DIM, 0, 1);
      int hh = k >> 4, e = k & 15;                    // B frag: e -> k=half*16+e
      v2bf p; p[0] = (__bf16)a0; p[1] = (__bf16)a1;
      *(v2bf*)&ldsB[((nn >> 4) << 9) + (((hh << 4) + (nn & 15)) << 4) + e] = p;
    }
    __syncthreads();

    v16bf afK = *(const v16bf*)&ldsA[(mi << 9) + (lane << 4)];
    v16bf afV = *(const v16bf*)&ldsA[((4 + mi) << 9) + (lane << 4)];
#pragma unroll
    for (int ni = 0; ni < 4; ni++) {
      v16bf bfr = *(const v16bf*)&ldsB[((nb + ni) << 9) + (lane << 4)];
      accK[ni] = wmma_bf16(afK, bfr, accK[ni]);
      accV[ni] = wmma_bf16(afV, bfr, accV[ni]);
    }
    __syncthreads();
  }

  // epilogue: D elem r -> row m = half*8+r, col n = lane%16
#pragma unroll
  for (int t = 0; t < 4; t++) {
    const int n = l0 + ((nb + t) << 4) + ln;
#pragma unroll
    for (int r = 0; r < 8; r++) {
      int m = m0 + (mi << 4) + (half << 3) + r;
      size_t off = XB + (size_t)m * L_DIM + n;
      OutK[off] = (__bf16)(accK[t][r] + bk[m]);
      OutV[off] = (__bf16)(accV[t][r] + bv[m]);
    }
  }
}

// ---------------------------------------------------------------------------
// Kernel 1b: single projection (queries).  Same tiling, one weight matrix.
// ---------------------------------------------------------------------------
__global__ __launch_bounds__(256) void proj_kernel(
    const float* __restrict__ W, const float* __restrict__ bias,
    const float* __restrict__ X, __bf16* __restrict__ Out)
{
  __shared__ __attribute__((aligned(32))) __bf16 ldsA[4 * 512];
  __shared__ __attribute__((aligned(32))) __bf16 ldsB[8 * 512];

  const int tid  = threadIdx.x;
  const int wave = tid >> 5, lane = tid & 31;
  const int half = lane >> 4, ln = lane & 15;
  const int mi = wave & 3;
  const int nb = (wave >> 2) << 2;
  const int m0 = blockIdx.y * 64;
  const int l0 = blockIdx.x * 128;
  const size_t XB = (size_t)blockIdx.z * C_DIM * L_DIM;

  v8f acc[4] = {{}, {}, {}, {}};

  for (int k0 = 0; k0 < C_DIM; k0 += 32) {
#pragma unroll
    for (int i = 0; i < 4; i++) {
      int idx = tid + i * 256;                        // 1024 pairs
      int k2 = idx & 15, m = idx >> 4;
      v2f w2 = *(const v2f*)&W[(size_t)(m0 + m) * C_DIM + (k0 + 2 * k2)];
      int k  = 2 * k2;
      int hh = k >> 4;
      int e  = k & 15;
      v2bf p; p[0] = (__bf16)w2[0]; p[1] = (__bf16)w2[1];
      *(v2bf*)&ldsA[((m >> 4) << 9) + (((hh << 4) + (m & 15)) << 4) + e] = p;
    }
#pragma unroll
    for (int i = 0; i < 8; i++) {
      int idx = tid + i * 256;                        // 2048 pairs
      int nn = idx & 127, k2 = idx >> 7;
      int k  = 2 * k2;
      const float* xp = &X[XB + (size_t)(k0 + k) * L_DIM + (l0 + nn)];
      float a0 = xp[0], a1 = xp[L_DIM];
      if (k0 + 32 < C_DIM) __builtin_prefetch(xp + 32 * L_DIM, 0, 1);
      int hh = k >> 4, e = k & 15;
      v2bf p; p[0] = (__bf16)a0; p[1] = (__bf16)a1;
      *(v2bf*)&ldsB[((nn >> 4) << 9) + (((hh << 4) + (nn & 15)) << 4) + e] = p;
    }
    __syncthreads();

    v16bf af = *(const v16bf*)&ldsA[(mi << 9) + (lane << 4)];
#pragma unroll
    for (int ni = 0; ni < 4; ni++) {
      v16bf bfr = *(const v16bf*)&ldsB[((nb + ni) << 9) + (lane << 4)];
      acc[ni] = wmma_bf16(af, bfr, acc[ni]);
    }
    __syncthreads();
  }

#pragma unroll
  for (int t = 0; t < 4; t++) {
    const int n = l0 + ((nb + t) << 4) + ln;
#pragma unroll
    for (int r = 0; r < 8; r++) {
      int m = m0 + (mi << 4) + (half << 3) + r;
      float v = acc[t][r] + bias[m];
      Out[XB + (size_t)m * L_DIM + n] = (__bf16)v;
    }
  }
}

// ---------------------------------------------------------------------------
// Kernel 2: softmax over tokens (L) for keys, in place. One block per (b,ch).
// Online (max,sum) per thread, LDS log-sum-exp tree combine.
// ---------------------------------------------------------------------------
__global__ __launch_bounds__(256) void key_softmax_kernel(__bf16* __restrict__ K)
{
  __bf16* p = K + (size_t)blockIdx.x * L_DIM;
  float m = -3.0e38f, s = 0.f;
  for (int i = threadIdx.x; i < L_DIM; i += 256) {
    float v = (float)p[i];
    if (v > m) { s = s * __expf(m - v) + 1.f; m = v; }
    else       { s += __expf(v - m); }
  }
  __shared__ float sm[256], ss[256];
  sm[threadIdx.x] = m; ss[threadIdx.x] = s;
  for (int off = 128; off > 0; off >>= 1) {
    __syncthreads();
    if (threadIdx.x < off) {
      float m1 = sm[threadIdx.x], s1 = ss[threadIdx.x];
      float m2 = sm[threadIdx.x + off], s2 = ss[threadIdx.x + off];
      float M = fmaxf(m1, m2);
      sm[threadIdx.x] = M;
      ss[threadIdx.x] = s1 * __expf(m1 - M) + s2 * __expf(m2 - M);
    }
  }
  __syncthreads();
  float M = sm[0], rS = 1.f / ss[0];
  // in-place safe: each element re-read and overwritten by the same thread
  for (int i = threadIdx.x; i < L_DIM; i += 256)
    p[i] = (__bf16)(__expf((float)p[i] - M) * rS);
}

// ---------------------------------------------------------------------------
// Kernel 3: softmax over the 32 head-channels for queries, in place.
// One thread per (b,h,l) column; adjacent threads -> adjacent l (coalesced).
// ---------------------------------------------------------------------------
__global__ __launch_bounds__(256) void query_softmax_kernel(__bf16* __restrict__ Q)
{
  int idx = blockIdx.x * 256 + threadIdx.x;            // n*HEADS*L total
  if (idx >= N_BATCH * NHEADS * L_DIM) return;
  int l  = idx % L_DIM;
  int bh = idx / L_DIM;                                // b*8+h ; b*C+h*32 == bh*32
  __bf16* p = Q + (size_t)bh * 32 * L_DIM + l;
  float m = -3.0e38f;
#pragma unroll
  for (int k = 0; k < 32; k++) m = fmaxf(m, (float)p[(size_t)k * L_DIM]);
  float ex[32]; float s = 0.f;
#pragma unroll
  for (int k = 0; k < 32; k++) { ex[k] = __expf((float)p[(size_t)k * L_DIM] - m); s += ex[k]; }
  float rS = 1.f / s;
#pragma unroll
  for (int k = 0; k < 32; k++) p[(size_t)k * L_DIM] = (__bf16)(ex[k] * rS);
}

// ---------------------------------------------------------------------------
// Kernel 4: ctx[b,h,i,j] = sum_l ksm[b,h*32+i,l] * V[b,h*32+j,l]
// One block per (b,h); K=9216 split over 8 waves (1152 tokens each),
// fragments loaded directly from global (bf16 rows contiguous),
// cross-wave f32 reduction in LDS. Output bf16 row-major 32x32.
// ---------------------------------------------------------------------------
__global__ __launch_bounds__(256) void ctx_kernel(
    const __bf16* __restrict__ ksm, const __bf16* __restrict__ V,
    __bf16* __restrict__ ctx)
{
  const int b = blockIdx.x / NHEADS, h = blockIdx.x % NHEADS;
  const int wave = threadIdx.x >> 5, lane = threadIdx.x & 31;
  const int half = lane >> 4, ln = lane & 15;
  const __bf16* Kb = ksm + ((size_t)b * C_DIM + h * 32) * L_DIM;
  const __bf16* Vb = V   + ((size_t)b * C_DIM + h * 32) * L_DIM;

  v8f acc[2][2] = {{{}, {}}, {{}, {}}};
  const int tpw = L_DIM / 8;                           // 1152 tokens per wave
  for (int t0 = wave * tpw; t0 < (wave + 1) * tpw; t0 += 32) {
    v16bf a[2], bb[2];
#pragma unroll
    for (int miq = 0; miq < 2; miq++) {
      const __bf16* rp = Kb + (size_t)(miq * 16 + ln) * L_DIM + t0;
      v8bf lo = *(const v8bf*)(rp + half * 16);        // k = half*16 + e
      v8bf hi = *(const v8bf*)(rp + half * 16 + 8);    // k = half*16 + 8 + (e-8)
      v16bf av;
#pragma unroll
      for (int e = 0; e < 8; e++) { av[e] = lo[e]; av[e + 8] = hi[e]; }
      a[miq] = av;
    }
#pragma unroll
    for (int ni = 0; ni < 2; ni++)                     // B: col n=lane%16, k=half*16+e
      bb[ni] = *(const v16bf*)(Vb + (size_t)(ni * 16 + ln) * L_DIM + t0 + half * 16);
#pragma unroll
    for (int miq = 0; miq < 2; miq++)
#pragma unroll
      for (int ni = 0; ni < 2; ni++)
        acc[miq][ni] = wmma_bf16(a[miq], bb[ni], acc[miq][ni]);
  }

  __shared__ float red[8 * 1024];                      // 32 KB
  float* myred = red + wave * 1024;
#pragma unroll
  for (int miq = 0; miq < 2; miq++)
#pragma unroll
    for (int ni = 0; ni < 2; ni++)
#pragma unroll
      for (int r = 0; r < 8; r++) {
        int i = miq * 16 + half * 8 + r;               // k-channel
        int j = ni * 16 + ln;                          // v-channel
        myred[i * 32 + j] = acc[miq][ni][r];
      }
  __syncthreads();
  for (int e = threadIdx.x; e < 1024; e += 256) {
    float s = 0.f;
#pragma unroll
    for (int w = 0; w < 8; w++) s += red[w * 1024 + e];
    ctx[(size_t)(b * NHEADS + h) * 1024 + e] = (__bf16)s;
  }
}

// ---------------------------------------------------------------------------
// Kernel 5: Weff[b][o, h*32+k] = sum_v Wr[o, h*32+v] * ctx[b,h,k,v]
// Folds the tiny per-head context into the output projection weight,
// eliminating the full-size "attended" tensor entirely.
// One block per (b,h): GEMM M=256, N=32, K=32 -> 4 WMMA per wave.
// ---------------------------------------------------------------------------
__global__ __launch_bounds__(256) void weff_kernel(
    const float* __restrict__ Wr, const __bf16* __restrict__ ctx,
    __bf16* __restrict__ Weff)
{
  const int b = blockIdx.x / NHEADS, h = blockIdx.x % NHEADS;
  const int wave = threadIdx.x >> 5, lane = threadIdx.x & 31;
  const int half = lane >> 4, ln = lane & 15;
  const __bf16* cb = ctx + (size_t)(b * NHEADS + h) * 1024;   // [k][v] 32x32

#pragma unroll
  for (int miq = 0; miq < 2; miq++) {
    const int m0 = wave * 32 + miq * 16;
    // A frag: A[m][v] = Wr[m, h*32+v], m = m0+ln, v per A layout
    const float* wp = Wr + (size_t)(m0 + ln) * C_DIM + h * 32;
    v16bf a;
#pragma unroll
    for (int e = 0; e < 8; e++) a[e]     = (__bf16)wp[half * 16 + e];
#pragma unroll
    for (int e = 0; e < 8; e++) a[e + 8] = (__bf16)wp[half * 16 + 8 + e];
#pragma unroll
    for (int ni = 0; ni < 2; ni++) {
      // B frag: B[v][k'] = ctx[k'][v]; col n = ni*16+ln, elem e -> v = half*16+e
      v16bf bf = *(const v16bf*)(cb + (size_t)(ni * 16 + ln) * 32 + half * 16);
      v8f acc = {};
      acc = wmma_bf16(a, bf, acc);
#pragma unroll
      for (int r = 0; r < 8; r++) {
        int o = m0 + half * 8 + r;
        int k = ni * 16 + ln;
        Weff[(size_t)b * C_DIM * C_DIM + (size_t)o * C_DIM + h * 32 + k] = (__bf16)acc[r];
      }
    }
  }
}

// ---------------------------------------------------------------------------
// Kernel 6: out[b,o,l] = sum_k Weff[b,o,k]*qsm[b,k,l] + br[o] + input_[b,o,l]
// Same 64x128 tiling; A is per-batch bf16, f32 output with residual add.
// grid = (L/128, C/64, n), block = 256
// ---------------------------------------------------------------------------
__global__ __launch_bounds__(256) void final_kernel(
    const __bf16* __restrict__ Weff, const __bf16* __restrict__ qsm,
    const float* __restrict__ br, const float* __restrict__ input_,
    float* __restrict__ out)
{
  __shared__ __attribute__((aligned(32))) __bf16 ldsA[4 * 512];
  __shared__ __attribute__((aligned(32))) __bf16 ldsB[8 * 512];

  const int tid  = threadIdx.x;
  const int wave = tid >> 5, lane = tid & 31;
  const int half = lane >> 4, ln = lane & 15;
  const int mi = wave & 3;
  const int nb = (wave >> 2) << 2;
  const int m0 = blockIdx.y * 64;
  const int l0 = blockIdx.x * 128;
  const size_t b  = blockIdx.z;
  const size_t XB = b * (size_t)C_DIM * L_DIM;
  const __bf16* Wb = Weff + b * (size_t)C_DIM * C_DIM;

  v8f acc[4] = {{}, {}, {}, {}};

  for (int k0 = 0; k0 < C_DIM; k0 += 32) {
    // --- A tile: bf16 pairs are contiguous in global, direct b32 copy ---
#pragma unroll
    for (int i = 0; i < 4; i++) {
      int idx = tid + i * 256;
      int k2 = idx & 15, m = idx >> 4;
      v2bf p = *(const v2bf*)&Wb[(size_t)(m0 + m) * C_DIM + (k0 + 2 * k2)];
      int k  = 2 * k2;
      int hh = k >> 4;
      int e  = k & 15;
      *(v2bf*)&ldsA[((m >> 4) << 9) + (((hh << 4) + (m & 15)) << 4) + e] = p;
    }
    // --- B tile: pack over adjacent k rows of qsm ---
#pragma unroll
    for (int i = 0; i < 8; i++) {
      int idx = tid + i * 256;
      int nn = idx & 127, k2 = idx >> 7;
      int k  = 2 * k2;
      const __bf16* qp = &qsm[XB + (size_t)(k0 + k) * L_DIM + (l0 + nn)];
      v2bf p; p[0] = qp[0]; p[1] = qp[L_DIM];
      if (k0 + 32 < C_DIM) __builtin_prefetch(qp + (size_t)32 * L_DIM, 0, 1);
      int hh = k >> 4, e = k & 15;
      *(v2bf*)&ldsB[((nn >> 4) << 9) + (((hh << 4) + (nn & 15)) << 4) + e] = p;
    }
    __syncthreads();

    v16bf af = *(const v16bf*)&ldsA[(mi << 9) + (lane << 4)];
#pragma unroll
    for (int ni = 0; ni < 4; ni++) {
      v16bf bfr = *(const v16bf*)&ldsB[((nb + ni) << 9) + (lane << 4)];
      acc[ni] = wmma_bf16(af, bfr, acc[ni]);
    }
    __syncthreads();
  }

#pragma unroll
  for (int t = 0; t < 4; t++) {
    const int n = l0 + ((nb + t) << 4) + ln;
#pragma unroll
    for (int r = 0; r < 8; r++) {
      int m = m0 + (mi << 4) + (half << 3) + r;
      size_t off = XB + (size_t)m * L_DIM + n;
      out[off] = acc[t][r] + br[m] + input_[off];
    }
  }
}

// ---------------------------------------------------------------------------
// Launch. Workspace layout (bytes), total ~229 MB:
//   Kb  bf16 n*C*L   (keys, softmaxed in place)
//   Qb  bf16 n*C*L   (queries, softmaxed in place)
//   Vb  bf16 n*C*L
//   ctx bf16 n*8*32*32
//   Weff bf16 n*C*C
// ---------------------------------------------------------------------------
extern "C" void kernel_launch(void* const* d_in, const int* in_sizes, int n_in,
                              void* d_out, int out_size, void* d_ws, size_t ws_size,
                              hipStream_t stream) {
  (void)in_sizes; (void)n_in; (void)out_size; (void)ws_size;
  const float* input_   = (const float*)d_in[0];
  const float* context_ = (const float*)d_in[1];
  const float* Wk = (const float*)d_in[2];
  const float* bk = (const float*)d_in[3];
  const float* Wq = (const float*)d_in[4];
  const float* bq = (const float*)d_in[5];
  const float* Wv = (const float*)d_in[6];
  const float* bv = (const float*)d_in[7];
  const float* Wr = (const float*)d_in[8];
  const float* br = (const float*)d_in[9];
  float* out = (float*)d_out;

  const size_t ncl = (size_t)N_BATCH * C_DIM * L_DIM;   // 37,748,736
  char* ws = (char*)d_ws;
  __bf16* Kb   = (__bf16*)ws;                 ws += ncl * 2;
  __bf16* Qb   = (__bf16*)ws;                 ws += ncl * 2;
  __bf16* Vb   = (__bf16*)ws;                 ws += ncl * 2;
  __bf16* ctxb = (__bf16*)ws;                 ws += (size_t)N_BATCH * NHEADS * 1024 * 2;
  __bf16* Weff = (__bf16*)ws;

  dim3 gproj(L_DIM / 128, C_DIM / 64, N_BATCH);         // (72, 4, 16)

  kv_proj_kernel<<<gproj, 256, 0, stream>>>(Wk, bk, Wv, bv, context_, Kb, Vb);
  proj_kernel<<<gproj, 256, 0, stream>>>(Wq, bq, input_, Qb);

  key_softmax_kernel<<<N_BATCH * C_DIM, 256, 0, stream>>>(Kb);
  query_softmax_kernel<<<(N_BATCH * NHEADS * L_DIM) / 256, 256, 0, stream>>>(Qb);

  ctx_kernel<<<N_BATCH * NHEADS, 256, 0, stream>>>(Kb, Vb, ctxb);
  weff_kernel<<<N_BATCH * NHEADS, 256, 0, stream>>>(Wr, ctxb, Weff);

  final_kernel<<<gproj, 256, 0, stream>>>(Weff, Qb, br, input_, out);
}
